// PointNetEmbedding_37168646979945
// MI455X (gfx1250) — matrix operand order repaired and checked by reading, not compiled, not run on hardware
//
#include <hip/hip_runtime.h>
#include <hip/hip_bf16.h>
#include <stdint.h>

typedef __attribute__((ext_vector_type(16))) __bf16 v16bf;
typedef __attribute__((ext_vector_type(8)))  __bf16 v8bf;
typedef __attribute__((ext_vector_type(8)))  float  v8f;

// ---------------------------------------------------------------------------
// Farthest point sampling: one block per batch cloud. min-distance array lives
// in LDS (<=8192 f32 = 32KB within the 320KB/WGP LDS budget).
// Tie-breaking matches jnp.argmax (first occurrence).
// ---------------------------------------------------------------------------
__global__ void fps_kernel(const float* __restrict__ pos, int Ns, int m,
                           int* __restrict__ idx, float* __restrict__ ctr) {
  const int b = blockIdx.x;
  const int t = threadIdx.x;
  const int nt = blockDim.x;
  __shared__ float mind[8192];
  __shared__ float rv[1024];
  __shared__ int   ri[1024];
  __shared__ float lp[3];
  const float* P = pos + (size_t)b * Ns * 3;
  for (int i = t; i < Ns; i += nt) mind[i] = 3.0e38f;
  if (t == 0) idx[(size_t)b * m] = 0;
  __syncthreads();
  int last = 0;
  for (int s = 1; s < m; ++s) {
    if (t == 0) { lp[0] = P[last*3]; lp[1] = P[last*3+1]; lp[2] = P[last*3+2]; }
    __syncthreads();
    const float cx = lp[0], cy = lp[1], cz = lp[2];
    float bv = -1.0f; int bi = 0;
    for (int i = t; i < Ns; i += nt) {
      float dx = P[i*3] - cx, dy = P[i*3+1] - cy, dz = P[i*3+2] - cz;
      float d = dx*dx + dy*dy + dz*dz;
      float mn = fminf(mind[i], d);
      mind[i] = mn;
      if (mn > bv) { bv = mn; bi = i; }   // strict >: keeps lowest index per thread
    }
    rv[t] = bv; ri[t] = bi;
    __syncthreads();
    for (int off = nt >> 1; off > 0; off >>= 1) {
      if (t < off) {
        float ov = rv[t+off]; int oi = ri[t+off];
        if (ov > rv[t] || (ov == rv[t] && oi < ri[t])) { rv[t] = ov; ri[t] = oi; }
      }
      __syncthreads();
    }
    last = ri[0];
    if (t == 0) idx[(size_t)b*m + s] = last;
    __syncthreads();
  }
  for (int j = t; j < m; j += nt) {
    int p = idx[(size_t)b*m + j];
    ctr[((size_t)b*m + j)*3 + 0] = P[p*3 + 0];
    ctr[((size_t)b*m + j)*3 + 1] = P[p*3 + 1];
    ctr[((size_t)b*m + j)*3 + 2] = P[p*3 + 2];
  }
}

// ---------------------------------------------------------------------------
// kNN (33 nearest) per center; distances resident in LDS, 33 iterative argmins
// (matches lax.top_k tie-order: smallest index first). Out-of-radius -> self.
// ---------------------------------------------------------------------------
__global__ void knn_kernel(const float* __restrict__ pos_src,
                           const float* __restrict__ ctr,
                           const int* __restrict__ cidx,
                           int Ns, int m, float r2, int* __restrict__ nbr) {
  const int c = blockIdx.x;
  const int b = c / m;
  const int t = threadIdx.x;
  __shared__ float d2s[8192];
  __shared__ float rv[256];
  __shared__ int   ri[256];
  __shared__ int   selj[33];
  __shared__ float seld[33];
  const float* P = pos_src + (size_t)b * Ns * 3;
  const float cx = ctr[(size_t)c*3], cy = ctr[(size_t)c*3+1], cz = ctr[(size_t)c*3+2];
  for (int i = t; i < Ns; i += 256) {
    float dx = P[i*3] - cx, dy = P[i*3+1] - cy, dz = P[i*3+2] - cz;
    d2s[i] = dx*dx + dy*dy + dz*dz;
  }
  __syncthreads();
  for (int k = 0; k < 33; ++k) {
    float bv = 3.0e38f; int bi = 0;
    for (int i = t; i < Ns; i += 256) {
      float v = d2s[i];
      if (v < bv) { bv = v; bi = i; }
    }
    rv[t] = bv; ri[t] = bi;
    __syncthreads();
    for (int off = 128; off > 0; off >>= 1) {
      if (t < off) {
        float ov = rv[t+off]; int oi = ri[t+off];
        if (ov < rv[t] || (ov == rv[t] && oi < ri[t])) { rv[t] = ov; ri[t] = oi; }
      }
      __syncthreads();
    }
    if (t == 0) { selj[k] = ri[0]; seld[k] = rv[0]; d2s[ri[0]] = 3.0e38f; }
    __syncthreads();
  }
  if (t < 33) nbr[(size_t)c*33 + t] = (seld[t] <= r2) ? selj[t] : cidx[c];
}

// ---------------------------------------------------------------------------
// Gather + concat [x_j, pos_j - ctr] into a K-padded bf16 activation matrix.
// ---------------------------------------------------------------------------
__global__ void build_feats_kernel(const float* __restrict__ xsrc, int F,
                                   const float* __restrict__ pos_src,
                                   const float* __restrict__ ctr,
                                   const int* __restrict__ nbr,
                                   int m, int Ns, __bf16* __restrict__ out, int ldk) {
  const int row = blockIdx.x;
  const int c = row / 33;
  const int b = c / m;
  const int j = nbr[row];
  const size_t src = (size_t)b * Ns + j;
  for (int col = threadIdx.x; col < ldk; col += blockDim.x) {
    float v;
    if (col < F)          v = xsrc[src * F + col];
    else if (col < F + 3) v = pos_src[src*3 + (col - F)] - ctr[(size_t)c*3 + (col - F)];
    else                  v = 0.0f;
    out[(size_t)row * ldk + col] = (__bf16)v;
  }
}

__global__ void concat_xpos_kernel(const float* __restrict__ x, int F,
                                   const float* __restrict__ p,
                                   __bf16* __restrict__ out, int ldk) {
  const int row = blockIdx.x;
  for (int col = threadIdx.x; col < ldk; col += blockDim.x) {
    float v;
    if (col < F)          v = x[(size_t)row*F + col];
    else if (col < F + 3) v = p[(size_t)row*3 + (col - F)];
    else                  v = 0.0f;
    out[(size_t)row*ldk + col] = (__bf16)v;
  }
}

// ---------------------------------------------------------------------------
// Weight prep: fp32 W[Kact,N] -> bf16, transposed + K-zero-padded Wt[N][Kpad].
// Makes the GEMM B-fragment two contiguous 16B loads (no divergence).
// ---------------------------------------------------------------------------
__global__ void prep_w_kernel(const float* __restrict__ W, int Kact, int N,
                              __bf16* __restrict__ Wt, int Kpad) {
  const int i = blockIdx.x * blockDim.x + threadIdx.x;
  if (i >= N * Kpad) return;
  const int n = i / Kpad;
  const int k = i % Kpad;
  Wt[i] = (__bf16)((k < Kact) ? W[(size_t)k * N + n] : 0.0f);
}

// ---------------------------------------------------------------------------
// WMMA bf16 GEMM: D = A(bf16 [M,Kpad]) * Wt^T (bf16 [N,Kpad]) + bias.
// One 16x16 tile per wave32, 8 waves/block. Fragment layouts per CDNA5 ISA
// 7.12.2. Inner loop is software-pipelined: next K-step's 4x b128 loads are
// issued before the current WMMA. flags: bit0 = ReLU, bit1 = fp32 store.
// ---------------------------------------------------------------------------
__global__ void wmma_gemm_bf16(const __bf16* __restrict__ A, int lda,
                               const __bf16* __restrict__ Wt,
                               const float* __restrict__ bias,
                               int M, int N, int Kpad,
                               void* __restrict__ out, int ldo, int flags) {
  const int lane  = threadIdx.x & 31;
  const int wid   = threadIdx.x >> 5;
  const int ntile = (blockIdx.x * 8 + wid) * 16;
  const int mtile = blockIdx.y * 16;
  if (ntile >= N) return;
  const int half = lane >> 4;     // 0: lanes 0-15, 1: lanes 16-31
  const int l15  = lane & 15;

  v8f acc;
#pragma unroll
  for (int i = 0; i < 8; ++i) acc[i] = 0.0f;

  const int ncol = ntile + l15;
  const __bf16* arow = A  + (size_t)(mtile + l15) * lda + half * 8;   // A: K {0..7}/{8..15} halves
  const __bf16* wrow = Wt + (size_t)ncol * Kpad        + half * 8;    // B: K {0..7}/{8..15} halves

  v8bf alo = *(const v8bf*)(arow);
  v8bf ahi = *(const v8bf*)(arow + 16);
  v8bf blo = *(const v8bf*)(wrow);
  v8bf bhi = *(const v8bf*)(wrow + 16);

  for (int k0 = 0; k0 + 32 < Kpad; k0 += 32) {
    v8bf alo2 = *(const v8bf*)(arow + k0 + 32);
    v8bf ahi2 = *(const v8bf*)(arow + k0 + 48);
    v8bf blo2 = *(const v8bf*)(wrow + k0 + 32);
    v8bf bhi2 = *(const v8bf*)(wrow + k0 + 48);
    v16bf af, bf;
#pragma unroll
    for (int e = 0; e < 8; ++e) {
      af[e] = alo[e]; af[8 + e] = ahi[e];
      bf[e] = blo[e]; bf[8 + e] = bhi[e];
    }
    acc = __builtin_amdgcn_wmma_f32_16x16x32_bf16(false, af, false, bf,
                                                  (short)0, acc, false, false);
    alo = alo2; ahi = ahi2; blo = blo2; bhi = bhi2;
  }
  {
    v16bf af, bf;
#pragma unroll
    for (int e = 0; e < 8; ++e) {
      af[e] = alo[e]; af[8 + e] = ahi[e];
      bf[e] = blo[e]; bf[8 + e] = bhi[e];
    }
    acc = __builtin_amdgcn_wmma_f32_16x16x32_bf16(false, af, false, bf,
                                                  (short)0, acc, false, false);
  }

  const float bv = bias ? bias[ncol] : 0.0f;
#pragma unroll
  for (int r = 0; r < 8; ++r) {
    const int mrow = mtile + r + half * 8;   // C/D layout: VGPR r -> M=r (+8 upper lanes)
    if (mrow < M) {
      float v = acc[r] + bv;
      if (flags & 1) v = fmaxf(v, 0.0f);
      if (flags & 2) ((float*)out)[(size_t)mrow * ldo + ncol] = v;
      else ((__bf16*)out)[(size_t)mrow * ldo + ncol] = (__bf16)v;
    }
  }
}

// Deterministic per-channel sum / sumsq (one block per channel; no atomics).
__global__ void bn_stats_kernel(const __bf16* __restrict__ X, int R, int C,
                                float* __restrict__ stats) {
  const int ch = blockIdx.x;
  const int t = threadIdx.x;
  float s = 0.f, s2 = 0.f;
  for (int r = t; r < R; r += 256) {
    float v = (float)X[(size_t)r * C + ch];
    s += v; s2 += v * v;
  }
  __shared__ float sh[256], sh2[256];
  sh[t] = s; sh2[t] = s2;
  __syncthreads();
  for (int off = 128; off > 0; off >>= 1) {
    if (t < off) { sh[t] += sh[t+off]; sh2[t] += sh2[t+off]; }
    __syncthreads();
  }
  if (t == 0) { stats[ch] = sh[0]; stats[C + ch] = sh2[0]; }
}

// In-place BN (train stats, biased var) + ReLU on bf16 tensor.
__global__ void bn_apply_kernel(__bf16* __restrict__ X, int R, int C,
                                const float* __restrict__ stats,
                                const float* __restrict__ g,
                                const float* __restrict__ be) {
  const size_t i = (size_t)blockIdx.x * blockDim.x + threadIdx.x;
  if (i >= (size_t)R * C) return;
  const int ch = (int)(i % C);
  const float invR = 1.0f / (float)R;
  const float mu = stats[ch] * invR;
  const float var = stats[C + ch] * invR - mu * mu;
  float v = (float)X[i];
  v = (v - mu) * rsqrtf(var + 1e-5f) * g[ch] + be[ch];
  X[i] = (__bf16)fmaxf(v, 0.0f);
}

// Max over the 33 neighbor messages of each center.
__global__ void maxagg_kernel(const __bf16* __restrict__ in, int C, int G,
                              __bf16* __restrict__ out) {
  const size_t i = (size_t)blockIdx.x * blockDim.x + threadIdx.x;
  if (i >= (size_t)G * C) return;
  const int gI = (int)(i / C), ch = (int)(i % C);
  float v = -3.0e38f;
  for (int k = 0; k < 33; ++k)
    v = fmaxf(v, (float)in[((size_t)gI * 33 + k) * C + ch]);
  out[i] = (__bf16)v;
}

// Global max pool over mg points per batch -> 16-row (zero padded) bf16 matrix.
__global__ void maxpool_kernel(const __bf16* __restrict__ in, int mg, int C, int Bv,
                               __bf16* __restrict__ out) {
  const int i = blockIdx.x * blockDim.x + threadIdx.x;
  if (i >= 16 * C) return;
  const int r = i / C, ch = i % C;
  float v = 0.0f;
  if (r < Bv) {
    v = -3.0e38f;
    for (int k = 0; k < mg; ++k)
      v = fmaxf(v, (float)in[((size_t)r * mg + k) * C + ch]);
  }
  out[i] = (__bf16)v;
}

// ---------------------------------------------------------------------------
extern "C" void kernel_launch(void* const* d_in, const int* in_sizes, int n_in,
                              void* d_out, int out_size, void* d_ws, size_t ws_size,
                              hipStream_t stream) {
  (void)in_sizes; (void)out_size;
  if (n_in < 51) return;

  constexpr int Bc = 4, Np = 8192, Kn = 33;
  constexpr int M1 = 2048, M2 = 512, M3 = 128;
  constexpr int R1 = Bc * M1 * Kn;   // 270336
  constexpr int R2 = Bc * M2 * Kn;   // 67584
  constexpr int R3 = Bc * M3 * Kn;   // 16896
  constexpr size_t BIG = (size_t)R1 * 128 * sizeof(__bf16);  // 69.2 MB

  char* ws = (char*)d_ws;
  size_t off = 0;
  auto carve = [&](size_t bytes) {
    char* p = ws + off;
    off = (off + bytes + 255) & ~(size_t)255;
    return (void*)p;
  };
  __bf16* BA   = (__bf16*)carve(BIG);
  __bf16* BB   = (__bf16*)carve(BIG);
  __bf16* WT   = (__bf16*)carve((size_t)1024 * 512 * 2 * 2);  // max N*Kpad, 2MB
  int*    IDX1 = (int*)  carve((size_t)Bc*M1*4);
  float*  CTR1 = (float*)carve((size_t)Bc*M1*3*4);
  int*    NBR1 = (int*)  carve((size_t)R1*4);
  float*  X1   = (float*)carve((size_t)Bc*M1*128*4);
  int*    IDX2 = (int*)  carve((size_t)Bc*M2*4);
  float*  CTR2 = (float*)carve((size_t)Bc*M2*3*4);
  int*    NBR2 = (int*)  carve((size_t)R2*4);
  float*  X2   = (float*)carve((size_t)Bc*M2*256*4);
  int*    IDX3 = (int*)  carve((size_t)Bc*M3*4);
  float*  CTR3 = (float*)carve((size_t)Bc*M3*3*4);
  int*    NBR3 = (int*)  carve((size_t)R3*4);
  float*  X3   = (float*)carve((size_t)Bc*M3*512*4);
  float*  STATS= (float*)carve(2*1024*4);
  __bf16* GBUF = (__bf16*)carve((size_t)16*1024*2);
  __bf16* F1b  = (__bf16*)carve((size_t)16*512*2);
  __bf16* F2b  = (__bf16*)carve((size_t)16*256*2);
  if (off > ws_size) return;  // insufficient scratch — bail deterministically

  const float* x0   = (const float*)d_in[0];
  const float* pos0 = (const float*)d_in[1];
  auto Wp = [&](int i) { return (const float*)d_in[i]; };

  // pytree flatten: dict keys sorted -> per layer order is W, b, be, g.
  auto gemm = [&](const __bf16* A, int lda, int wi, int bi,
                  int M, int N, int Kact, int Kpad, void* out, int ldo, int flags) {
    int wtot = N * Kpad;
    prep_w_kernel<<<(wtot + 255)/256, 256, 0, stream>>>(Wp(wi), Kact, N, WT, Kpad);
    dim3 grd((unsigned)((N/16 + 7)/8), (unsigned)((M + 15)/16));
    wmma_gemm_bf16<<<grd, 256, 0, stream>>>(A, lda, WT,
                                            bi >= 0 ? Wp(bi) : nullptr,
                                            M, N, Kpad, out, ldo, flags);
  };
  auto bn = [&](__bf16* X, int R, int C, int gi, int bei) {
    bn_stats_kernel<<<C, 256, 0, stream>>>(X, R, C, STATS);
    size_t tot = (size_t)R * C;
    bn_apply_kernel<<<(unsigned)((tot + 255)/256), 256, 0, stream>>>(
        X, R, C, STATS, Wp(gi), Wp(bei));
  };
  auto magg = [&](const __bf16* in, int C, int G, __bf16* out) {
    size_t tot = (size_t)G * C;
    maxagg_kernel<<<(unsigned)((tot + 255)/256), 256, 0, stream>>>(in, C, G, out);
  };

  // ----- SA module 1: N=8192 -> m=2048, r2=0.04, local [6,64,64,128], global [128,128]
  fps_kernel<<<Bc, 1024, 0, stream>>>(pos0, Np, M1, IDX1, CTR1);
  knn_kernel<<<Bc*M1, 256, 0, stream>>>(pos0, CTR1, IDX1, Np, M1, 0.04f, NBR1);
  build_feats_kernel<<<R1, 64, 0, stream>>>(x0, 3, pos0, CTR1, NBR1, M1, Np, BA, 32);
  gemm(BA, 32, 3, 4,  R1, 64, 6, 32,  BB, 64, 0);   bn(BB, R1, 64, 6, 5);
  gemm(BB, 64, 7, 8,  R1, 64, 64, 64, BA, 64, 0);   bn(BA, R1, 64, 10, 9);
  gemm(BA, 64, 11, 12, R1, 128, 64, 64, BB, 128, 0);          // plain last
  magg(BB, 128, Bc*M1, BA);
  gemm(BA, 128, 13, 14, Bc*M1, 128, 128, 128, X1, 128, 2);    // global MLP -> f32

  // ----- SA module 2: 2048 -> 512, r2=0.16, local [131,128,128,256], global [256,256]
  fps_kernel<<<Bc, 1024, 0, stream>>>(CTR1, M1, M2, IDX2, CTR2);
  knn_kernel<<<Bc*M2, 256, 0, stream>>>(CTR1, CTR2, IDX2, M1, M2, 0.16f, NBR2);
  build_feats_kernel<<<R2, 64, 0, stream>>>(X1, 128, CTR1, CTR2, NBR2, M2, M1, BA, 160);
  gemm(BA, 160, 15, 16, R2, 128, 131, 160, BB, 128, 0);  bn(BB, R2, 128, 18, 17);
  gemm(BB, 128, 19, 20, R2, 128, 128, 128, BA, 128, 0);  bn(BA, R2, 128, 22, 21);
  gemm(BA, 128, 23, 24, R2, 256, 128, 128, BB, 256, 0);
  magg(BB, 256, Bc*M2, BA);
  gemm(BA, 256, 25, 26, Bc*M2, 256, 256, 256, X2, 256, 2);

  // ----- SA module 3: 512 -> 128, r2=0.64, local [259,256,256,512], global [512,512]
  fps_kernel<<<Bc, 1024, 0, stream>>>(CTR2, M2, M3, IDX3, CTR3);
  knn_kernel<<<Bc*M3, 256, 0, stream>>>(CTR2, CTR3, IDX3, M2, M3, 0.64f, NBR3);
  build_feats_kernel<<<R3, 64, 0, stream>>>(X2, 256, CTR2, CTR3, NBR3, M3, M2, BA, 288);
  gemm(BA, 288, 27, 28, R3, 256, 259, 288, BB, 256, 0);  bn(BB, R3, 256, 30, 29);
  gemm(BB, 256, 31, 32, R3, 256, 256, 256, BA, 256, 0);  bn(BA, R3, 256, 34, 33);
  gemm(BA, 256, 35, 36, R3, 512, 256, 256, BB, 512, 0);
  magg(BB, 512, Bc*M3, BA);
  gemm(BA, 512, 37, 38, Bc*M3, 512, 512, 512, X3, 512, 2);

  // ----- sa4: concat [x(512), pos(3)] -> [515,512,1024] (BN+ReLU then plain), max pool
  concat_xpos_kernel<<<Bc*M3, 64, 0, stream>>>(X3, 512, CTR3, BA, 544);
  gemm(BA, 544, 39, 40, Bc*M3, 512, 515, 544, BB, 512, 0);  bn(BB, Bc*M3, 512, 42, 41);
  gemm(BB, 512, 43, 44, Bc*M3, 1024, 512, 512, BA, 1024, 0);
  maxpool_kernel<<<(16*1024 + 255)/256, 256, 0, stream>>>(BA, M3, 1024, Bc, GBUF);

  // ----- final head: [1024,512,256,128], no norm; ReLU on first two, plain last -> f32
  gemm(GBUF, 1024, 45, 46, Bc, 512, 1024, 1024, F1b, 512, 1);
  gemm(F1b,  512,  47, 48, Bc, 256, 512, 512,  F2b, 256, 1);
  gemm(F2b,  256,  49, 50, Bc, 128, 256, 256,  d_out, 128, 2);
}
